// Mamba_8632884265482
// MI455X (gfx1250) — compile-verified, matrix operand order checked by reading, not run on hardware
//
#include <hip/hip_runtime.h>
#include <math.h>

// ---------------------------------------------------------------------------
// Mamba-style block stack on gfx1250 (MI455X).
// bf16 WMMA GEMMs (double-buffered async global->LDS staging) + VALU scan.
// ---------------------------------------------------------------------------

typedef __bf16 bf16;
typedef __attribute__((ext_vector_type(16))) __bf16 v16bf;
typedef __attribute__((ext_vector_type(8)))  float  v8f;
typedef __attribute__((ext_vector_type(4)))  int    v4i;

#if defined(__HIP_DEVICE_COMPILE__)
#if __has_builtin(__builtin_amdgcn_global_load_async_to_lds_b128) && \
    __has_builtin(__builtin_amdgcn_s_wait_asynccnt)
#define ATHENA_ASYNC_LDS 1
typedef __attribute__((address_space(1))) v4i gv4i_t;  // global 128-bit chunk
typedef __attribute__((address_space(3))) v4i lv4i_t;  // LDS 128-bit chunk
#endif
#endif

// ------------------------------ small kernels ------------------------------

__global__ void embed_kernel(const int* __restrict__ tokens,
                             const float* __restrict__ embed,
                             float* __restrict__ out, int D) {
  int row = blockIdx.x;
  int tok = tokens[row];
  for (int d = threadIdx.x; d < D; d += blockDim.x)
    out[(size_t)row * D + d] = embed[(size_t)tok * D + d];
}

__global__ __launch_bounds__(256) void rmsnorm_kernel(
    const float* __restrict__ in, const float* __restrict__ gamma,
    float* __restrict__ out, bf16* __restrict__ outbf, int D) {
  int row = blockIdx.x;
  const float* x = in + (size_t)row * D;
  __shared__ float red[8];
  float ss = 0.f;
  for (int d = threadIdx.x; d < D; d += 256) { float v = x[d]; ss += v * v; }
  for (int off = 16; off > 0; off >>= 1) ss += __shfl_down(ss, off, 32);
  int lane = threadIdx.x & 31, wv = threadIdx.x >> 5;
  if (lane == 0) red[wv] = ss;
  __syncthreads();
  if (wv == 0) {
    float v = (lane < 8) ? red[lane] : 0.f;
    for (int off = 4; off > 0; off >>= 1) v += __shfl_down(v, off, 32);
    if (lane == 0) red[0] = v;
  }
  __syncthreads();
  float norm = sqrtf(red[0] / (float)D);
  float scale = gamma[0] / (norm + 1e-5f);
  for (int d = threadIdx.x; d < D; d += 256) {
    float v = x[d] * scale;
    out[(size_t)row * D + d] = v;
    if (outbf) outbf[(size_t)row * D + d] = (bf16)v;
  }
}

// strided fp32 -> bf16 convert (activations; row-major preserved)
__global__ void cvt_bf16_kernel(const float* __restrict__ in,
                                bf16* __restrict__ out,
                                long long total, int cols,
                                long long istride, long long ioff) {
  long long i = (long long)blockIdx.x * 256 + threadIdx.x;
  if (i >= total) return;
  long long r = i / cols, c = i % cols;
  out[i] = (bf16)in[r * istride + ioff + c];
}

// weight fp32 [K][N] row-major -> bf16 transposed [Npad][K], zero pad rows.
// 32x32 LDS tile transpose, 256 threads as 32x8.
__global__ __launch_bounds__(256) void cvt_transpose_kernel(
    const float* __restrict__ in, bf16* __restrict__ out, int K, int N,
    int Npad) {
  __shared__ float tile[32][33];
  int n0 = blockIdx.x * 32, k0 = blockIdx.y * 32;
  int tx = threadIdx.x & 31, ty = threadIdx.x >> 5;  // 32 x 8
#pragma unroll
  for (int r = 0; r < 32; r += 8) {
    int k = k0 + ty + r, n = n0 + tx;
    tile[ty + r][tx] = (k < K && n < N) ? in[(size_t)k * N + n] : 0.f;
  }
  __syncthreads();
#pragma unroll
  for (int r = 0; r < 32; r += 8) {
    int n = n0 + ty + r, k = k0 + tx;
    if (n < Npad && k < K) out[(size_t)n * K + k] = (bf16)tile[tx][ty + r];
  }
}

// depthwise conv1d (k=3, same pad) + bias + sigmoid; u taken from z[:, :DH]
__global__ void conv_sigmoid_kernel(const float* __restrict__ z,
                                    const float* __restrict__ cw,
                                    const float* __restrict__ cb,
                                    float* __restrict__ u,
                                    bf16* __restrict__ ub,
                                    int Bn, int L, int DH) {
  long long idx = (long long)blockIdx.x * 256 + threadIdx.x;
  long long total = (long long)Bn * L * DH;
  if (idx >= total) return;
  int dh = (int)(idx % DH);
  int l  = (int)((idx / DH) % L);
  int b  = (int)(idx / ((long long)DH * L));
  const int ZS = 2 * DH;
  float w0 = cw[dh * 3 + 0], w1 = cw[dh * 3 + 1], w2 = cw[dh * 3 + 2];
  float s = cb[dh];
  const float* zb = z + (size_t)b * L * ZS + dh;
  if (l - 1 >= 0) s += zb[(size_t)(l - 1) * ZS] * w0;
  s += zb[(size_t)l * ZS] * w1;
  if (l + 1 < L) s += zb[(size_t)(l + 1) * ZS] * w2;
  float v = 1.f / (1.f + __expf(-s));
  u[idx] = v;
  ub[idx] = (bf16)v;
}

// SSM linear recurrence: one lane per (batch, channel); N=8 states.
__global__ void ssm_scan_kernel(const float* __restrict__ proj,
                                const float* __restrict__ u,
                                const float* __restrict__ Acoef,
                                const float* __restrict__ Dp,
                                float* __restrict__ h,
                                int Bn, int L, int DH) {
  const int Nst = 8;
  int idx = blockIdx.x * 256 + threadIdx.x;
  if (idx >= Bn * DH) return;
  int dh = idx % DH, b = idx / DH;
  const int PS = (2 * Nst + 1) * DH;
  float s[Nst], Ac[Nst];
#pragma unroll
  for (int n = 0; n < Nst; ++n) { s[n] = 0.f; Ac[n] = Acoef[dh * Nst + n]; }
  float dp = Dp[dh];
  const float* pb = proj + (size_t)b * L * PS;
  const float* ubp = u + (size_t)b * L * DH + dh;
  float* hb = h + (size_t)b * L * DH + dh;
  for (int t = 0; t < L; ++t) {
    const float* pr = pb + (size_t)t * PS;
    float delta = pr[2 * Nst * DH + dh];
    float uv = ubp[(size_t)t * DH];
    float du = delta * uv;
    float y = 0.f;
#pragma unroll
    for (int n = 0; n < Nst; ++n) {
      float ab = __expf(-delta * Ac[n]);
      float bu = du * pr[dh * Nst + n];
      s[n] = ab * s[n] + bu;
      y += pr[Nst * DH + dh * Nst + n] * s[n];
    }
    hb[(size_t)t * DH] = y + dp * uv;
  }
}

// ------------------------------ WMMA GEMM ----------------------------------
// C[M,N](fp32) = A[M,K](bf16, row-major) @ B (given as BT[Npad][K] bf16,
// transposed + zero padded to Npad = multiple of 128) + bias.
// Epilogues: mode 0: +bias | mode 1: gelu(sigmoid(acc+bias)*H) | mode 2:
// +bias+X residual. Optional bf16 mirror Cbf. Double-buffered async staging.
#define TM 128
#define TN 128
#define TK 32
#define LSTR 40  // padded LDS stride (elements)
#define TILE_ELEMS (TM * LSTR)

__global__ __launch_bounds__(256) void gemm_bf16_wmma(
    const bf16* __restrict__ A, const bf16* __restrict__ BT,
    const float* __restrict__ bias, const float* __restrict__ H,
    const float* __restrict__ X, float* __restrict__ C,
    bf16* __restrict__ Cbf, int M, int N, int K, int mode) {
  __shared__ __align__(16) bf16 As[2 * TILE_ELEMS];  // [buf][m][k]
  __shared__ __align__(16) bf16 Bs[2 * TILE_ELEMS];  // [buf][n][k]

  const int tid  = threadIdx.x;
  const int lane = tid & 31;
  const int wave = tid >> 5;
  const int wm = wave >> 2;   // 0..1 -> 64 rows each
  const int wn = wave & 3;    // 0..3 -> 32 cols each
  const int lm = lane & 15;
  const int kh = lane >> 4;

  const int m0 = blockIdx.y * TM;
  const int n0 = blockIdx.x * TN;

  v8f acc[4][2];
#pragma unroll
  for (int i = 0; i < 4; ++i)
#pragma unroll
    for (int j = 0; j < 2; ++j)
#pragma unroll
      for (int r = 0; r < 8; ++r) acc[i][j][r] = 0.f;

  // staging map: 256 threads -> 128 rows x (2 x 16 bf16) = 128x32 tile
  const int srow = tid >> 1;
  const int scol = (tid & 1) * 16;
  const bf16* agp = A  + (size_t)(m0 + srow) * K + scol;
  const bf16* bgp = BT + (size_t)(n0 + srow) * K + scol;
  bf16* alp = &As[srow * LSTR + scol];
  bf16* blp = &Bs[srow * LSTR + scol];

  auto stage = [&](int buf, int k0) {
#ifdef ATHENA_ASYNC_LDS
    __builtin_amdgcn_global_load_async_to_lds_b128(
        (gv4i_t*)(agp + k0), (lv4i_t*)(alp + buf * TILE_ELEMS), 0, 0);
    __builtin_amdgcn_global_load_async_to_lds_b128(
        (gv4i_t*)(agp + k0 + 8), (lv4i_t*)(alp + buf * TILE_ELEMS + 8), 0, 0);
    __builtin_amdgcn_global_load_async_to_lds_b128(
        (gv4i_t*)(bgp + k0), (lv4i_t*)(blp + buf * TILE_ELEMS), 0, 0);
    __builtin_amdgcn_global_load_async_to_lds_b128(
        (gv4i_t*)(bgp + k0 + 8), (lv4i_t*)(blp + buf * TILE_ELEMS + 8), 0, 0);
#else
    uint4 a0 = *(const uint4*)(agp + k0);
    uint4 a1 = *(const uint4*)(agp + k0 + 8);
    uint4 b0 = *(const uint4*)(bgp + k0);
    uint4 b1 = *(const uint4*)(bgp + k0 + 8);
    *(uint4*)(alp + buf * TILE_ELEMS)     = a0;
    *(uint4*)(alp + buf * TILE_ELEMS + 8) = a1;
    *(uint4*)(blp + buf * TILE_ELEMS)     = b0;
    *(uint4*)(blp + buf * TILE_ELEMS + 8) = b1;
#endif
  };

  stage(0, 0);
#ifdef ATHENA_ASYNC_LDS
  __builtin_amdgcn_s_wait_asynccnt(0);
#endif
  __syncthreads();

  const int nk = K / TK;
  for (int kt = 0; kt < nk; ++kt) {
    const int cur = kt & 1;
    if (kt + 1 < nk) stage(cur ^ 1, (kt + 1) * TK);

    const bf16* Ab = &As[cur * TILE_ELEMS];
    const bf16* Bb = &Bs[cur * TILE_ELEMS];
    // A layout per lane (16-bit 16x32): lanes 0-15 hold K {0..7,16..23},
    // lanes 16-31 hold K {8..15,24..31}, row M = lane&15. BT identical on N.
    v16bf af[4];
#pragma unroll
    for (int i = 0; i < 4; ++i) {
      const bf16* p = Ab + (wm * 64 + i * 16 + lm) * LSTR;
      union { uint4 u[2]; v16bf v; } t;
      t.u[0] = *(const uint4*)(p + kh * 8);
      t.u[1] = *(const uint4*)(p + 16 + kh * 8);
      af[i] = t.v;
    }
    v16bf bfr[2];
#pragma unroll
    for (int j = 0; j < 2; ++j) {
      const bf16* p = Bb + (wn * 32 + j * 16 + lm) * LSTR;
      union { uint4 u[2]; v16bf v; } t;
      t.u[0] = *(const uint4*)(p + kh * 8);
      t.u[1] = *(const uint4*)(p + 16 + kh * 8);
      bfr[j] = t.v;
    }
#pragma unroll
    for (int i = 0; i < 4; ++i)
#pragma unroll
      for (int j = 0; j < 2; ++j)
        acc[i][j] = __builtin_amdgcn_wmma_f32_16x16x32_bf16(
            false, af[i], false, bfr[j], (short)0, acc[i][j], false, false);

#ifdef ATHENA_ASYNC_LDS
    __builtin_amdgcn_s_wait_asynccnt(0);
#endif
    __syncthreads();
  }

  // ---- epilogue ----
#pragma unroll
  for (int i = 0; i < 4; ++i) {
#pragma unroll
    for (int j = 0; j < 2; ++j) {
      int gn = n0 + wn * 32 + j * 16 + lm;
      if (gn >= N) continue;
      float bv = bias ? bias[gn] : 0.f;
#pragma unroll
      for (int r = 0; r < 8; ++r) {
        int gm = m0 + wm * 64 + i * 16 + kh * 8 + r;
        size_t idx = (size_t)gm * N + gn;
        float v = acc[i][j][r] + bv;
        if (mode == 1) {
          float s = 1.f / (1.f + __expf(-v));
          float t = s * H[idx];
          v = 0.5f * t * (1.f + erff(t * 0.70710678118654752f));
        } else if (mode == 2) {
          v += X[idx];
        }
        C[idx] = v;
        if (Cbf) Cbf[idx] = (bf16)v;
      }
    }
  }
}

// B fragment wants BT; note wave-pair (wm) reuses the same B columns, giving
// LDS-level reuse; weights stream through L2 (192 MB holds the full bf16 set).

// ------------------------------ host driver --------------------------------

extern "C" void kernel_launch(void* const* d_in, const int* in_sizes, int n_in,
                              void* d_out, int out_size, void* d_ws,
                              size_t ws_size, hipStream_t stream) {
  (void)in_sizes; (void)n_in; (void)out_size; (void)ws_size;
  const int*   tokens    = (const int*)d_in[0];
  const float* embed     = (const float*)d_in[1];
  const float* W_in_w    = (const float*)d_in[2];
  const float* W_in_b    = (const float*)d_in[3];
  const float* conv_w    = (const float*)d_in[4];
  const float* conv_b    = (const float*)d_in[5];
  const float* W_g_w     = (const float*)d_in[6];
  const float* W_g_b     = (const float*)d_in[7];
  const float* x_proj_w  = (const float*)d_in[8];
  const float* x_proj_b  = (const float*)d_in[9];
  const float* W_out_w   = (const float*)d_in[10];
  const float* W_out_b   = (const float*)d_in[11];
  const float* Acoef     = (const float*)d_in[12];
  const float* Dparam    = (const float*)d_in[13];
  const float* gamma_in  = (const float*)d_in[14];
  const float* gamma_out = (const float*)d_in[15];
  const float* fgamma    = (const float*)d_in[16];
  const float* unembed_w = (const float*)d_in[17];
  const float* unembed_b = (const float*)d_in[18];
  float* out = (float*)d_out;

  constexpr int Bn = 2, L = 1024, D = 512, DH = 1024, Nst = 8, V = 10000,
                DEPTH = 4;
  constexpr int M = Bn * L;              // 2048 token rows
  constexpr int P = (2 * Nst + 1) * DH;  // 17408
  constexpr int Vpad = 10112;            // V rounded up to 128

  char* wsp = (char*)d_ws;
  auto alloc = [&](size_t bytes) -> void* {
    void* p = (void*)wsp;
    wsp += (bytes + 255) & ~(size_t)255;
    return p;
  };
  float* state = (float*)alloc((size_t)M * D * 4);
  float* xn    = (float*)alloc((size_t)M * D * 4);
  bf16*  xnb   = (bf16*)alloc((size_t)M * D * 2);
  float* z     = (float*)alloc((size_t)M * 2 * DH * 4);
  float* u     = (float*)alloc((size_t)M * DH * 4);
  bf16*  ub    = (bf16*)alloc((size_t)M * DH * 2);
  bf16*  gb    = (bf16*)alloc((size_t)M * DH * 2);
  float* proj  = (float*)alloc((size_t)M * P * 4);
  float* h     = (float*)alloc((size_t)M * DH * 4);
  float* y     = (float*)alloc((size_t)M * DH * 4);
  bf16*  yb    = (bf16*)alloc((size_t)M * DH * 2);
  float* xres  = (float*)alloc((size_t)M * D * 4);
  bf16*  wb    = (bf16*)alloc((size_t)DH * P * 2);  // transposed weight buf

  auto cvt = [&](const float* src, bf16* dst, long long total, int cols,
                 long long istride, long long ioff) {
    int blocks = (int)((total + 255) / 256);
    cvt_bf16_kernel<<<blocks, 256, 0, stream>>>(src, dst, total, cols, istride,
                                                ioff);
  };
  // weight [K][N] fp32 -> [Npad][K] bf16 (zero padded)
  auto cvtT = [&](const float* src, bf16* dst, int K, int N, int Npad) {
    dim3 grid(Npad / 32, K / 32);
    cvt_transpose_kernel<<<grid, 256, 0, stream>>>(src, dst, K, N, Npad);
  };
  auto gemm = [&](const bf16* Am, const bf16* BTm, const float* bias,
                  const float* Hm, const float* Xm, float* Cm, bf16* Cb,
                  int Mm, int Nn, int Kk, int mode) {
    dim3 grid((Nn + TN - 1) / TN, Mm / TM);
    gemm_bf16_wmma<<<grid, 256, 0, stream>>>(Am, BTm, bias, Hm, Xm, Cm, Cb,
                                             Mm, Nn, Kk, mode);
  };

  embed_kernel<<<M, 256, 0, stream>>>(tokens, embed, state, D);

  for (int lyr = 0; lyr < DEPTH; ++lyr) {
    // x = rmsnorm(x, gamma_in)
    rmsnorm_kernel<<<M, 256, 0, stream>>>(state, gamma_in + lyr, xn, xnb, D);
    // z = x @ Wi + bi
    cvtT(W_in_w + (size_t)lyr * D * 2 * DH, wb, D, 2 * DH, 2 * DH);
    gemm(xnb, wb, W_in_b + (size_t)lyr * 2 * DH, nullptr, nullptr, z, nullptr,
         M, 2 * DH, D, 0);
    // u = sigmoid(conv1d(z[:, :DH]) + cb)
    {
      long long total = (long long)Bn * L * DH;
      conv_sigmoid_kernel<<<(int)((total + 255) / 256), 256, 0, stream>>>(
          z, conv_w + (size_t)lyr * DH * 3, conv_b + (size_t)lyr * DH, u, ub,
          Bn, L, DH);
    }
    // g (bf16) from z[:, DH:]
    cvt(z, gb, (long long)M * DH, DH, 2 * DH, DH);
    // proj = u @ Wx + bx
    cvtT(x_proj_w + (size_t)lyr * DH * P, wb, DH, P, P);
    gemm(ub, wb, x_proj_b + (size_t)lyr * P, nullptr, nullptr, proj, nullptr,
         M, P, DH, 0);
    // sequential SSM scan -> h
    ssm_scan_kernel<<<(Bn * DH + 255) / 256, 256, 0, stream>>>(
        proj, u, Acoef + (size_t)lyr * DH * Nst, Dparam + (size_t)lyr * DH, h,
        Bn, L, DH);
    // y = gelu(sigmoid(g @ Wg + bg) * h)
    cvtT(W_g_w + (size_t)lyr * DH * DH, wb, DH, DH, DH);
    gemm(gb, wb, W_g_b + (size_t)lyr * DH, h, nullptr, y, yb, M, DH, DH, 1);
    // xres = y @ Wo + bo + xn  (residual with normed input, per reference)
    cvtT(W_out_w + (size_t)lyr * DH * D, wb, DH, D, D);
    gemm(yb, wb, W_out_b + (size_t)lyr * D, nullptr, xn, xres, nullptr, M, D,
         DH, 2);
    // x = rmsnorm(xres, gamma_out)
    rmsnorm_kernel<<<M, 256, 0, stream>>>(xres, gamma_out + lyr, state,
                                          nullptr, D);
  }

  // final norm + unembed
  rmsnorm_kernel<<<M, 256, 0, stream>>>(state, fgamma, xn, xnb, D);
  cvtT(unembed_w, wb, D, V, Vpad);
  gemm(xnb, wb, unembed_b, nullptr, nullptr, out, nullptr, M, V, D, 0);
}
